// MultiHeadAttention_4604204941382
// MI455X (gfx1250) — matrix operand order repaired and checked by reading, not compiled, not run on hardware
//
#include <hip/hip_runtime.h>
#include <hip/hip_bf16.h>

typedef __bf16 bf16;
typedef __attribute__((ext_vector_type(16))) __bf16 bf16x16;
typedef __attribute__((ext_vector_type(8)))  __bf16 bf16x8;
typedef __attribute__((ext_vector_type(4)))  __bf16 bf16x4;
typedef __attribute__((ext_vector_type(8)))  float  f32x8;

#define LDA 40     // 32 + 8 bf16 pad -> 80B row pitch, 16B aligned, spreads banks
#define TBUF (128 * LDA)   // one LDS tile buffer (elements)

// ---------------- CDNA5 async LDS copy (ASYNCcnt path), with safe fallback ----
#if defined(__has_builtin)
#if __has_builtin(__builtin_amdgcn_global_load_async_to_lds_b128)
#define HAS_ASYNC_LDS 1
#endif
#endif
#ifndef HAS_ASYNC_LDS
#define HAS_ASYNC_LDS 0
#endif

#if HAS_ASYNC_LDS
typedef __attribute__((__vector_size__(16))) int v4i;
typedef __attribute__((address_space(1))) v4i gv4i;   // global int32x4
typedef __attribute__((address_space(3))) v4i lv4i;   // LDS int32x4
__device__ __forceinline__ void async_cp16(const bf16* g, bf16* l) {
  // 16B per lane, global -> LDS, no VGPR staging (tracked by ASYNCcnt)
  __builtin_amdgcn_global_load_async_to_lds_b128((gv4i*)g, (lv4i*)l, 0, 0);
}
#if __has_builtin(__builtin_amdgcn_s_wait_asynccnt)
#define WAIT_ASYNC(n) __builtin_amdgcn_s_wait_asynccnt(n)
#else
#define WAIT_ASYNC(n) asm volatile("s_wait_asynccnt %0" ::"n"(n) : "memory")
#endif
#endif

__device__ __forceinline__ bf16x16 frag_cat(bf16x8 lo, bf16x8 hi) {
  return __builtin_shufflevector(lo, hi, 0,1,2,3,4,5,6,7,8,9,10,11,12,13,14,15);
}

// A fragment (16x32 bf16, ISA 7.12.2): lane L<16 -> row L, K {0..7,16..23};
// lane L>=16 -> row L-16, K {8..15,24..31}
__device__ __forceinline__ bf16x16 load_fragA(const bf16* t, int lane) {
  int m  = lane & 15;
  int ko = (lane >> 4) << 3;               // 0 or 8
  const bf16* p = t + m * LDA + ko;
  return frag_cat(*(const bf16x8*)p, *(const bf16x8*)(p + 16));
}

// B fragment (32x16 bf16), tile stored N-major (Bt[n][k]):
// lanes 0-15 hold K {0..7,16..23} of column n=lane; lanes 16-31 hold K {8..15,24..31}
__device__ __forceinline__ bf16x16 load_fragB(const bf16* t, int lane) {
  int n  = lane & 15;
  int kb = (lane >> 4) << 3;               // 0 or 8
  const bf16* p = t + n * LDA + kb;
  return frag_cat(*(const bf16x8*)p, *(const bf16x8*)(p + 16));
}

#define WMMA_BF16(a,b,c) \
  __builtin_amdgcn_wmma_f32_16x16x32_bf16(false,(a),false,(b),(short)0,(c),false,false)

// ---------------------------------------------------------------- converts

__global__ __launch_bounds__(256) void cvt_f32_bf16(const float* __restrict__ in,
                                                    bf16* __restrict__ out, int n4) {
  int i = blockIdx.x * 256 + threadIdx.x;
  if (i >= n4) return;
  float4 f = ((const float4*)in)[i];
  bf16x4 o = {(bf16)f.x, (bf16)f.y, (bf16)f.z, (bf16)f.w};
  ((bf16x4*)out)[i] = o;
}

// out[c][r] = (bf16) in[r][c]   (in: R x C f32, out: C x R bf16)
__global__ __launch_bounds__(256) void transpose_cvt(const float* __restrict__ in,
                                                     bf16* __restrict__ out, int R, int C) {
  __shared__ float tile[32][33];
  int tx = threadIdx.x & 31, ty = threadIdx.x >> 5;     // 32 x 8
  int r0 = blockIdx.y * 32, c0 = blockIdx.x * 32;
  #pragma unroll
  for (int i = 0; i < 32; i += 8) {
    int r = r0 + ty + i, c = c0 + tx;
    tile[ty + i][tx] = (r < R && c < C) ? in[(size_t)r * C + c] : 0.f;
  }
  __syncthreads();
  #pragma unroll
  for (int i = 0; i < 32; i += 8) {
    int rOut = c0 + ty + i, cOut = r0 + tx;
    if (rOut < C && cOut < R)
      out[(size_t)rOut * R + cOut] = (bf16)tile[tx][ty + i];
  }
}

// ------------------------------------------------- QKV projection GEMM
// C[8192,2304] = Xb[8192,768] @ Wqkv + b ; scatter -> Q[b,h,s,d], K[b,h,s,d], Vt[b,h,d,s]
__global__ __launch_bounds__(256) void qkv_gemm(const bf16* __restrict__ X,
                                                const bf16* __restrict__ Wt,
                                                const float* __restrict__ bias,
                                                bf16* __restrict__ Q,
                                                bf16* __restrict__ Kd,
                                                bf16* __restrict__ Vt) {
  __shared__ __align__(16) bf16 As[2 * TBUF];
  __shared__ __align__(16) bf16 Bs[2 * TBUF];
  int tid = threadIdx.x, lane = tid & 31, wv = tid >> 5;
  int waveM = wv & 3, waveN = wv >> 2;
  int m0 = blockIdx.y * 128, n0 = blockIdx.x * 128;
  int lrow = tid >> 1, lcol = (tid & 1) * 16;

  const bf16* gA = X  + (size_t)(m0 + lrow) * 768 + lcol;
  const bf16* gB = Wt + (size_t)(n0 + lrow) * 768 + lcol;
  bf16* lA = As + lrow * LDA + lcol;
  bf16* lB = Bs + lrow * LDA + lcol;

  f32x8 acc[2][4] = {};

#if HAS_ASYNC_LDS
  // prologue: k=0 tile into buffer 0
  async_cp16(gA,     lA);
  async_cp16(gA + 8, lA + 8);
  async_cp16(gB,     lB);
  async_cp16(gB + 8, lB + 8);
  for (int k0 = 0; k0 < 768; k0 += 32) {
    int cur = (k0 >> 5) & 1, nxt = cur ^ 1;
    if (k0 + 32 < 768) {
      async_cp16(gA + k0 + 32,     lA + nxt * TBUF);
      async_cp16(gA + k0 + 32 + 8, lA + nxt * TBUF + 8);
      async_cp16(gB + k0 + 32,     lB + nxt * TBUF);
      async_cp16(gB + k0 + 32 + 8, lB + nxt * TBUF + 8);
      WAIT_ASYNC(4);     // in-order completion: current buffer is resident
    } else {
      WAIT_ASYNC(0);
    }
    __syncthreads();
    bf16x16 af[2], bf[4];
    #pragma unroll
    for (int i = 0; i < 2; i++)
      af[i] = load_fragA(As + cur * TBUF + (32 * waveM + 16 * i) * LDA, lane);
    #pragma unroll
    for (int j = 0; j < 4; j++)
      bf[j] = load_fragB(Bs + cur * TBUF + (64 * waveN + 16 * j) * LDA, lane);
    #pragma unroll
    for (int i = 0; i < 2; i++)
      #pragma unroll
      for (int j = 0; j < 4; j++)
        acc[i][j] = WMMA_BF16(af[i], bf[j], acc[i][j]);
    __syncthreads();   // done reading 'cur' before it is refilled at k0+64
  }
#else
  for (int k0 = 0; k0 < 768; k0 += 32) {
    const bf16x8* sa = (const bf16x8*)(gA + k0);
    const bf16x8* sb = (const bf16x8*)(gB + k0);
    bf16x8 a0 = sa[0], a1 = sa[1], b0 = sb[0], b1 = sb[1];
    if (k0 + 32 < 768) {
      __builtin_prefetch(gA + k0 + 32, 0, 3);
      __builtin_prefetch(gB + k0 + 32, 0, 3);
    }
    *(bf16x8*)(lA)     = a0;
    *(bf16x8*)(lA + 8) = a1;
    *(bf16x8*)(lB)     = b0;
    *(bf16x8*)(lB + 8) = b1;
    __syncthreads();
    bf16x16 af[2], bf[4];
    #pragma unroll
    for (int i = 0; i < 2; i++) af[i] = load_fragA(As + (32 * waveM + 16 * i) * LDA, lane);
    #pragma unroll
    for (int j = 0; j < 4; j++) bf[j] = load_fragB(Bs + (64 * waveN + 16 * j) * LDA, lane);
    #pragma unroll
    for (int i = 0; i < 2; i++)
      #pragma unroll
      for (int j = 0; j < 4; j++)
        acc[i][j] = WMMA_BF16(af[i], bf[j], acc[i][j]);
    __syncthreads();
  }
#endif

  int col0 = lane & 15, row0 = (lane >> 4) * 8;
  #pragma unroll
  for (int i = 0; i < 2; i++)
    #pragma unroll
    for (int j = 0; j < 4; j++) {
      int gn = n0 + 64 * waveN + 16 * j + col0;
      int c = gn % 3, rem = gn / 3;
      int d = rem & 63, h = rem >> 6;
      float bv = bias[gn];
      #pragma unroll
      for (int r = 0; r < 8; r++) {
        int gm = m0 + 32 * waveM + 16 * i + row0 + r;
        int b = gm >> 10, s = gm & 1023;
        bf16 o = (bf16)(acc[i][j][r] + bv);
        size_t bh = (size_t)(b * 12 + h);
        if (c == 0)      Q [(bh * 1024 + s) * 64 + d] = o;
        else if (c == 1) Kd[(bh * 1024 + s) * 64 + d] = o;
        else             Vt[(bh * 64 + d) * 1024 + s] = o;
      }
    }
}

// ------------------------------------------------- scores = scale * Q @ K^T (f32 out)
__global__ __launch_bounds__(256) void scores_gemm(const bf16* __restrict__ Q,
                                                   const bf16* __restrict__ Kd,
                                                   float* __restrict__ Wts) {
  __shared__ __align__(16) bf16 As[128 * LDA];
  __shared__ __align__(16) bf16 Bs[128 * LDA];
  int bh = blockIdx.z;
  const bf16* A = Q  + (size_t)bh * 1024 * 64;
  const bf16* B = Kd + (size_t)bh * 1024 * 64;
  float* W = Wts + ((size_t)bh << 20);
  int tid = threadIdx.x, lane = tid & 31, wv = tid >> 5;
  int waveM = wv & 3, waveN = wv >> 2;
  int m0 = blockIdx.y * 128, n0 = blockIdx.x * 128;
  int lrow = tid >> 1, lcol = (tid & 1) * 16;

  f32x8 acc[2][4] = {};
  #pragma unroll
  for (int k0 = 0; k0 < 64; k0 += 32) {
    const bf16x8* sa = (const bf16x8*)(A + (size_t)(m0 + lrow) * 64 + k0 + lcol);
    const bf16x8* sb = (const bf16x8*)(B + (size_t)(n0 + lrow) * 64 + k0 + lcol);
    bf16x8 a0 = sa[0], a1 = sa[1], b0 = sb[0], b1 = sb[1];
    *(bf16x8*)(As + lrow * LDA + lcol)     = a0;
    *(bf16x8*)(As + lrow * LDA + lcol + 8) = a1;
    *(bf16x8*)(Bs + lrow * LDA + lcol)     = b0;
    *(bf16x8*)(Bs + lrow * LDA + lcol + 8) = b1;
    __syncthreads();
    bf16x16 af[2], bf[4];
    #pragma unroll
    for (int i = 0; i < 2; i++) af[i] = load_fragA(As + (32 * waveM + 16 * i) * LDA, lane);
    #pragma unroll
    for (int j = 0; j < 4; j++) bf[j] = load_fragB(Bs + (64 * waveN + 16 * j) * LDA, lane);
    #pragma unroll
    for (int i = 0; i < 2; i++)
      #pragma unroll
      for (int j = 0; j < 4; j++)
        acc[i][j] = WMMA_BF16(af[i], bf[j], acc[i][j]);
    __syncthreads();
  }
  int col0 = lane & 15, row0 = (lane >> 4) * 8;
  #pragma unroll
  for (int i = 0; i < 2; i++)
    #pragma unroll
    for (int j = 0; j < 4; j++) {
      int gn = n0 + 64 * waveN + 16 * j + col0;
      #pragma unroll
      for (int r = 0; r < 8; r++) {
        int gm = m0 + 32 * waveM + 16 * i + row0 + r;
        W[(size_t)gm * 1024 + gn] = 0.125f * acc[i][j][r];
      }
    }
}

// ------------------------------------------------- row softmax in place (f32)
__global__ __launch_bounds__(256) void softmax_rows(float* __restrict__ Wts) {
  __shared__ float red[256];
  float* p = Wts + (size_t)blockIdx.x * 1024;
  int tid = threadIdx.x;
  float4 v = ((float4*)p)[tid];
  float m = fmaxf(fmaxf(v.x, v.y), fmaxf(v.z, v.w));
  red[tid] = m;
  __syncthreads();
  #pragma unroll
  for (int s = 128; s > 0; s >>= 1) {
    if (tid < s) red[tid] = fmaxf(red[tid], red[tid + s]);
    __syncthreads();
  }
  m = red[0];
  __syncthreads();
  float e0 = __expf(v.x - m), e1 = __expf(v.y - m);
  float e2 = __expf(v.z - m), e3 = __expf(v.w - m);
  red[tid] = e0 + e1 + e2 + e3;
  __syncthreads();
  #pragma unroll
  for (int s = 128; s > 0; s >>= 1) {
    if (tid < s) red[tid] += red[tid + s];
    __syncthreads();
  }
  float inv = 1.0f / red[0];
  float4 o; o.x = e0 * inv; o.y = e1 * inv; o.z = e2 * inv; o.w = e3 * inv;
  ((float4*)p)[tid] = o;
}

// ------------------------------------------------- attn = weights @ V  (bf16 out [B,S,H*D])
__global__ __launch_bounds__(256) void attn_gemm(const float* __restrict__ Wts,
                                                 const bf16* __restrict__ Vt,
                                                 bf16* __restrict__ out) {
  __shared__ __align__(16) bf16 As[128 * LDA];
  __shared__ __align__(16) bf16 Bs[64 * LDA];
  int bh = blockIdx.y;
  int b = bh / 12, h = bh % 12;
  const float* A = Wts + ((size_t)bh << 20);
  const bf16*  V = Vt + (size_t)bh * 64 * 1024;
  int tid = threadIdx.x, lane = tid & 31, wv = tid >> 5;
  int m0 = blockIdx.x * 128;
  int lrow = tid >> 1, lcol = (tid & 1) * 16;
  int lrB = tid >> 2, lcB = (tid & 3) * 8;

  f32x8 acc[4] = {};
  for (int k0 = 0; k0 < 1024; k0 += 32) {
    const float4* sa = (const float4*)(A + (size_t)(m0 + lrow) * 1024 + k0 + lcol);
    float4 f0 = sa[0], f1 = sa[1], f2 = sa[2], f3 = sa[3];
    bf16x8 vb = *(const bf16x8*)(V + (size_t)lrB * 1024 + k0 + lcB);
    if (k0 + 32 < 1024)
      __builtin_prefetch(A + (size_t)(m0 + lrow) * 1024 + k0 + 32 + lcol, 0, 3);
    bf16x8 lo = {(bf16)f0.x,(bf16)f0.y,(bf16)f0.z,(bf16)f0.w,
                 (bf16)f1.x,(bf16)f1.y,(bf16)f1.z,(bf16)f1.w};
    bf16x8 hi = {(bf16)f2.x,(bf16)f2.y,(bf16)f2.z,(bf16)f2.w,
                 (bf16)f3.x,(bf16)f3.y,(bf16)f3.z,(bf16)f3.w};
    *(bf16x8*)(As + lrow * LDA + lcol)     = lo;
    *(bf16x8*)(As + lrow * LDA + lcol + 8) = hi;
    *(bf16x8*)(Bs + lrB * LDA + lcB)       = vb;
    __syncthreads();
    bf16x16 af = load_fragA(As + 16 * wv * LDA, lane);
    bf16x16 bf[4];
    #pragma unroll
    for (int j = 0; j < 4; j++) bf[j] = load_fragB(Bs + 16 * j * LDA, lane);
    #pragma unroll
    for (int j = 0; j < 4; j++) acc[j] = WMMA_BF16(af, bf[j], acc[j]);
    __syncthreads();
  }
  int col0 = lane & 15, row0 = (lane >> 4) * 8;
  #pragma unroll
  for (int j = 0; j < 4; j++) {
    int d = 16 * j + col0;
    #pragma unroll
    for (int r = 0; r < 8; r++) {
      int s = m0 + 16 * wv + row0 + r;
      out[((size_t)b * 1024 + s) * 768 + h * 64 + d] = (bf16)acc[j][r];
    }
  }
}

// ------------------------------------------------- out = attn @ W_out + b (f32 out)
__global__ __launch_bounds__(256) void out_gemm(const bf16* __restrict__ X,
                                                const bf16* __restrict__ Wt,
                                                const float* __restrict__ bias,
                                                float* __restrict__ out) {
  __shared__ __align__(16) bf16 As[2 * TBUF];
  __shared__ __align__(16) bf16 Bs[2 * TBUF];
  int tid = threadIdx.x, lane = tid & 31, wv = tid >> 5;
  int waveM = wv & 3, waveN = wv >> 2;
  int m0 = blockIdx.y * 128, n0 = blockIdx.x * 128;
  int lrow = tid >> 1, lcol = (tid & 1) * 16;

  const bf16* gA = X  + (size_t)(m0 + lrow) * 768 + lcol;
  const bf16* gB = Wt + (size_t)(n0 + lrow) * 768 + lcol;
  bf16* lA = As + lrow * LDA + lcol;
  bf16* lB = Bs + lrow * LDA + lcol;

  f32x8 acc[2][4] = {};

#if HAS_ASYNC_LDS
  async_cp16(gA,     lA);
  async_cp16(gA + 8, lA + 8);
  async_cp16(gB,     lB);
  async_cp16(gB + 8, lB + 8);
  for (int k0 = 0; k0 < 768; k0 += 32) {
    int cur = (k0 >> 5) & 1, nxt = cur ^ 1;
    if (k0 + 32 < 768) {
      async_cp16(gA + k0 + 32,     lA + nxt * TBUF);
      async_cp16(gA + k0 + 32 + 8, lA + nxt * TBUF + 8);
      async_cp16(gB + k0 + 32,     lB + nxt * TBUF);
      async_cp16(gB + k0 + 32 + 8, lB + nxt * TBUF + 8);
      WAIT_ASYNC(4);
    } else {
      WAIT_ASYNC(0);
    }
    __syncthreads();
    bf16x16 af[2], bf[4];
    #pragma unroll
    for (int i = 0; i < 2; i++)
      af[i] = load_fragA(As + cur * TBUF + (32 * waveM + 16 * i) * LDA, lane);
    #pragma unroll
    for (int j = 0; j < 4; j++)
      bf[j] = load_fragB(Bs + cur * TBUF + (64 * waveN + 16 * j) * LDA, lane);
    #pragma unroll
    for (int i = 0; i < 2; i++)
      #pragma unroll
      for (int j = 0; j < 4; j++)
        acc[i][j] = WMMA_BF16(af[i], bf[j], acc[i][j]);
    __syncthreads();
  }
#else
  for (int k0 = 0; k0 < 768; k0 += 32) {
    const bf16x8* sa = (const bf16x8*)(gA + k0);
    const bf16x8* sb = (const bf16x8*)(gB + k0);
    bf16x8 a0 = sa[0], a1 = sa[1], b0 = sb[0], b1 = sb[1];
    if (k0 + 32 < 768)
      __builtin_prefetch(gA + k0 + 32, 0, 3);
    *(bf16x8*)(lA)     = a0;
    *(bf16x8*)(lA + 8) = a1;
    *(bf16x8*)(lB)     = b0;
    *(bf16x8*)(lB + 8) = b1;
    __syncthreads();
    bf16x16 af[2], bf[4];
    #pragma unroll
    for (int i = 0; i < 2; i++) af[i] = load_fragA(As + (32 * waveM + 16 * i) * LDA, lane);
    #pragma unroll
    for (int j = 0; j < 4; j++) bf[j] = load_fragB(Bs + (64 * waveN + 16 * j) * LDA, lane);
    #pragma unroll
    for (int i = 0; i < 2; i++)
      #pragma unroll
      for (int j = 0; j < 4; j++)
        acc[i][j] = WMMA_BF16(af[i], bf[j], acc[i][j]);
    __syncthreads();
  }
#endif

  int col0 = lane & 15, row0 = (lane >> 4) * 8;
  #pragma unroll
  for (int i = 0; i < 2; i++)
    #pragma unroll
    for (int j = 0; j < 4; j++) {
      int gn = n0 + 64 * waveN + 16 * j + col0;
      float bv = bias[gn];
      #pragma unroll
      for (int r = 0; r < 8; r++) {
        int gm = m0 + 32 * waveM + 16 * i + row0 + r;
        out[(size_t)gm * 768 + gn] = acc[i][j][r] + bv;
      }
    }
}

// ---------------------------------------------------------------- launch

extern "C" void kernel_launch(void* const* d_in, const int* in_sizes, int n_in,
                              void* d_out, int out_size, void* d_ws, size_t ws_size,
                              hipStream_t stream) {
  const float* x     = (const float*)d_in[0];   // [8,1024,768]
  const float* W_qkv = (const float*)d_in[1];   // [768,2304]
  const float* b_qkv = (const float*)d_in[2];   // [2304]
  const float* W_out = (const float*)d_in[3];   // [768,768]
  const float* b_out = (const float*)d_in[4];   // [768]

  float* out = (float*)d_out;                       // [8,1024,768]
  float* wts = out + (size_t)8 * 1024 * 768;        // [8,12,1024,1024]

  char* ws = (char*)d_ws;
  bf16* xb     = (bf16*)(ws);                        // 12,582,912 B
  bf16* wqkv_t = (bf16*)(ws + 12582912);             //  3,538,944 B  [2304][768]
  bf16* wout_t = (bf16*)(ws + 16121856);             //  1,179,648 B  [768][768]
  bf16* Q      = (bf16*)(ws + 17301504);             // 12,582,912 B  [96][1024][64]
  bf16* Kd     = (bf16*)(ws + 29884416);             // 12,582,912 B  [96][1024][64]
  bf16* Vt     = (bf16*)(ws + 42467328);             // 12,582,912 B  [96][64][1024]
  bf16* attnb  = (bf16*)(ws + 55050240);             // 12,582,912 B  [8192][768]

  cvt_f32_bf16<<<6144, 256, 0, stream>>>(x, xb, 1572864);
  transpose_cvt<<<dim3(72, 24), 256, 0, stream>>>(W_qkv, wqkv_t, 768, 2304);
  transpose_cvt<<<dim3(24, 24), 256, 0, stream>>>(W_out, wout_t, 768, 768);

  qkv_gemm<<<dim3(18, 64), 256, 0, stream>>>(xb, wqkv_t, b_qkv, Q, Kd, Vt);
  scores_gemm<<<dim3(8, 8, 96), 256, 0, stream>>>(Q, Kd, wts);
  softmax_rows<<<98304, 256, 0, stream>>>(wts);
  attn_gemm<<<dim3(8, 96), 256, 0, stream>>>(wts, Vt, attnb);
  out_gemm<<<dim3(6, 64), 256, 0, stream>>>(attnb, wout_t, b_out, out);
}
